// QLSTM_65481071397533
// MI455X (gfx1250) — matrix operand order repaired and checked by reading, not compiled, not run on hardware
//
#include <hip/hip_runtime.h>
#include <hip/hip_bf16.h>
#include <math.h>
#include <stdint.h>

// ---------------------------------------------------------------------------
// LSTM over SEQ=2048 steps, B=64, D=512, H=512 on gfx1250 (MI455X).
// Persistent-kernel design: 32 WGs x 256 threads; each WG owns 16 hidden
// units (=> 64 gate columns f/i/g/o). Per step: 64x64x1024 bf16 WMMA GEMM
// (A = [x_t, h] staged in LDS as bf16, B = pre-transcoded bf16 weights from
// L2), branch-free gate fusion (v_exp/v_rcp), device-scope arrive/wait
// between steps, async Global->LDS copy for the recurrent h operand.
// ---------------------------------------------------------------------------

typedef __attribute__((ext_vector_type(16))) __bf16 v16bf;
typedef __attribute__((ext_vector_type(4)))  __bf16 v4bf;
typedef __attribute__((ext_vector_type(4)))  float  v4f;
typedef __attribute__((ext_vector_type(8)))  float  v8f;

#define SEQ   2048
#define BATCH 64
#define DIM   512
#define HID   512
#define KTOT  1024          // DIM + HID
#define NCOL  2048          // 4 gates * HID columns
#define NWG   32            // 512 units / 16 units-per-WG
#define UPW   16            // hidden units per WG
#define NTHR  256
#define ASTR  1048          // LDS A row stride (bf16): 16B-aligned, conflict-free
#define GSTR  68            // LDS gate scratch row stride (floats)
#define HSZ   (BATCH*HID)   // elements in one h buffer

// workspace layout (bytes)
#define WS_COUNTERS 0                         // 2048 * 4B
#define WS_HBUF     8192                      // 2 * 64*512 * 2B = 131072
#define WS_WT       139264                    // 2048*1024*2B = 4194304
// total ws needed: 4,333,568 bytes

#define LOG2E  1.44269504088896340736f

union FragBF { uint4 q[2]; v16bf v; };

__device__ __forceinline__ float fast_sigmoid(float x) {
    // 1 / (1 + 2^(-x*log2e)); v_exp_f32 + v_rcp_f32, branch-free, saturating.
    return __builtin_amdgcn_rcpf(1.0f + __builtin_amdgcn_exp2f(-LOG2E * x));
}
__device__ __forceinline__ float fast_tanh(float x) {
    // 1 - 2/(2^(2x*log2e)+1); rcp(inf)=0 -> +/-1 at the tails.
    const float e2 = __builtin_amdgcn_exp2f(2.0f * LOG2E * x);
    return fmaf(-2.0f, __builtin_amdgcn_rcpf(e2 + 1.0f), 1.0f);
}

// --- prep 1: zero step counters + h0 buffer (contiguous region) -------------
extern "C" __global__ void lstm_zero_ws(uint4* __restrict__ p, int n4) {
    int i = blockIdx.x * blockDim.x + threadIdx.x;
    if (i < n4) p[i] = make_uint4(0u, 0u, 0u, 0u);
}

// --- prep 2: transcode gate weights fp32 -> bf16, concatenated [f,i,g,o] ----
// WT[(gate*512 + unit)][k] = W_gate[unit][k]; rows are K-contiguous, exactly
// the layout WMMA B-fragments want (per-column contiguous K).
extern "C" __global__ void lstm_convert_w(const float* __restrict__ Wf,
                                          const float* __restrict__ Wi,
                                          const float* __restrict__ Wg,
                                          const float* __restrict__ Wo,
                                          __bf16* __restrict__ WT) {
    const int i = blockIdx.x * blockDim.x + threadIdx.x;   // over float4 chunks
    const int total4 = (NCOL * KTOT) / 4;                  // 524288
    if (i >= total4) return;
    const int g = i >> 17;                                 // 131072 chunks / gate
    const int r = i & ((1 << 17) - 1);
    const float* src = (g == 0) ? Wf : (g == 1) ? Wi : (g == 2) ? Wg : Wo;
    const v4f v = ((const v4f*)src)[r];
    ((v4bf*)WT)[i] = __builtin_convertvector(v, v4bf);
}

// --- main persistent kernel -------------------------------------------------
extern "C" __global__ void __launch_bounds__(NTHR)
lstm_seq_kernel(const float* __restrict__ xin,     // [SEQ][64][512]
                const float* __restrict__ bfp, const float* __restrict__ bip,
                const float* __restrict__ bgp, const float* __restrict__ bop,
                const __bf16* __restrict__ WT,     // [2048][1024] bf16
                __bf16* __restrict__ hbuf,         // [2][64][512] bf16
                int*    __restrict__ counters,     // [SEQ]
                float*  __restrict__ out)          // outputs | hx | cx
{
    extern __shared__ char smem[];
    __bf16* As    = (__bf16*)smem;                                  // [64][ASTR]
    float*  gates = (float*)(smem + (size_t)BATCH * ASTR * sizeof(__bf16)); // [64][GSTR]

    const int wg   = blockIdx.x;       // 0..31 -> units [wg*16, wg*16+16)
    const int tid  = threadIdx.x;
    const int lane = tid & 31;
    const int wave = tid >> 5;         // 0..7
    const int nt   = wave & 3;         // gate tile (f,i,g,o)
    const int mt0  = (wave >> 2) << 1; // 0 or 2
    const int lr   = lane & 15;
    const int hi   = lane >> 4;

    // B fragment source row: column (nt*512 + wg*16 + lr) of the gate matrix,
    // K-contiguous. Lane reads 16B chunks at kb+hi*8, kb+16+hi*8 (matches A).
    const __bf16* Brow  = WT + (size_t)(nt * HID + wg * UPW + lr) * KTOT;
    // A fragment rows in LDS (16-bit A layout: chunks at kb+hi*8, kb+16+hi*8).
    const __bf16* Arow0 = As + (size_t)(mt0 * 16 + lr) * ASTR;
    const __bf16* Arow1 = Arow0 + 16 * ASTR;

    // elementwise mapping: thread -> (batch eb, 4 consecutive units)
    const int eb = tid & 63;
    const int eu = tid >> 6;           // 0..3
    const int ub = eu * 4;             // local unit base  [0,16)
    const int ug = wg * UPW + ub;      // global unit base

    float cst[4]   = {0.f, 0.f, 0.f, 0.f};
    float hlast[4] = {0.f, 0.f, 0.f, 0.f};
    float Bf4[4], Bi4[4], Bg4[4], Bo4[4];
#pragma unroll
    for (int j = 0; j < 4; ++j) {
        Bf4[j] = bfp[ug + j]; Bi4[j] = bip[ug + j];
        Bg4[j] = bgp[ug + j]; Bo4[j] = bop[ug + j];
    }

    for (int t = 0; t < SEQ; ++t) {
        // ---- wait until every WG published h for step t-1 ----
        if (t > 0) {
            if (tid == 0) {
                while (__hip_atomic_load(&counters[t - 1], __ATOMIC_RELAXED,
                                         __HIP_MEMORY_SCOPE_AGENT) != NWG)
                    __builtin_amdgcn_s_sleep(1);
            }
            __syncthreads();
            __builtin_amdgcn_fence(__ATOMIC_ACQUIRE, "agent");  // global_inv
        }

        // ---- stage A = [x_t | h] into LDS as bf16 ----
        // h half: async Global->LDS copy (16B per lane, ASYNCcnt-tracked).
        {
            const __bf16* hsrc = hbuf + (size_t)(t & 1) * HSZ;
            for (int i = tid; i < (BATCH * HID) / 8; i += NTHR) {
                const int b = i >> 6;              // / (HID/8)
                const int k = (i & 63) << 3;
                // LDS aperture: low 32 bits of the generic address are the
                // LDS byte offset (ISA 10.2 per-aperture address calc).
                const unsigned ldsOff =
                    (unsigned)(uintptr_t)(As + (size_t)b * ASTR + DIM + k);
                const unsigned long long gaddr =
                    (unsigned long long)(uintptr_t)(hsrc + (size_t)b * HID + k);
                asm volatile("global_load_async_to_lds_b128 %0, %1, off"
                             :: "v"(ldsOff), "v"(gaddr) : "memory");
            }
        }
        // x half: fp32 -> bf16 convert (packed cvt via convertvector).
        const float* xt = xin + (size_t)t * (BATCH * DIM);
        for (int i = tid; i < (BATCH * DIM) / 4; i += NTHR) {
            const int b = i >> 7;              // / (DIM/4)
            const int k = (i & 127) << 2;
            const v4f v = ((const v4f*)xt)[(size_t)b * (DIM / 4) + (k >> 2)];
            *((v4bf*)(As + (size_t)b * ASTR + k)) = __builtin_convertvector(v, v4bf);
        }
        if (t + 1 < SEQ) {                     // warm x_{t+1} (global_prefetch_b8)
            const float* nx = xin + (size_t)(t + 1) * (BATCH * DIM) + tid * 128;
            __builtin_prefetch(nx +  0, 0, 1);
            __builtin_prefetch(nx + 32, 0, 1);
            __builtin_prefetch(nx + 64, 0, 1);
            __builtin_prefetch(nx + 96, 0, 1);
        }
        asm volatile("s_wait_asynccnt 0x0" ::: "memory");
        __syncthreads();

        // ---- 64x64x1024 GEMM: 2 accumulator tiles per wave ----
        v8f acc0 = {0.f, 0.f, 0.f, 0.f, 0.f, 0.f, 0.f, 0.f};
        v8f acc1 = {0.f, 0.f, 0.f, 0.f, 0.f, 0.f, 0.f, 0.f};
#pragma unroll 4
        for (int kb = 0; kb < KTOT; kb += 32) {
            FragBF bfr, a0, a1;
            const __bf16* bp = Brow + kb + hi * 8;
            bfr.q[0] = *((const uint4*)(bp));
            bfr.q[1] = *((const uint4*)(bp + 16));
            const __bf16* ap0 = Arow0 + kb + hi * 8;
            a0.q[0] = *((const uint4*)(ap0));
            a0.q[1] = *((const uint4*)(ap0 + 16));
            const __bf16* ap1 = Arow1 + kb + hi * 8;
            a1.q[0] = *((const uint4*)(ap1));
            a1.q[1] = *((const uint4*)(ap1 + 16));
            acc0 = __builtin_amdgcn_wmma_f32_16x16x32_bf16(
                       false, a0.v, false, bfr.v, (short)0, acc0, false, false);
            acc1 = __builtin_amdgcn_wmma_f32_16x16x32_bf16(
                       false, a1.v, false, bfr.v, (short)0, acc1, false, false);
        }

        // ---- dump accumulators (C/D layout: row = r + 8*hi, col = lr) ----
        {
            float* gp0 = gates + (size_t)(mt0 * 16 + hi * 8) * GSTR + nt * 16 + lr;
            float* gp1 = gates + (size_t)(mt0 * 16 + 16 + hi * 8) * GSTR + nt * 16 + lr;
#pragma unroll
            for (int r = 0; r < 8; ++r) {
                gp0[(size_t)r * GSTR] = acc0[r];
                gp1[(size_t)r * GSTR] = acc1[r];
            }
        }
        __syncthreads();

        // ---- gate fusion; c kept in registers across all steps ----
        const float* grow = gates + (size_t)eb * GSTR + ub;
        v4f hvv;
#pragma unroll
        for (int j = 0; j < 4; ++j) {
            const float fv = fast_sigmoid(grow[j]      + Bf4[j]);
            const float iv = fast_sigmoid(grow[16 + j] + Bi4[j]);
            const float gv = fast_tanh   (grow[32 + j] + Bg4[j]);
            const float ov = fast_sigmoid(grow[48 + j] + Bo4[j]);
            cst[j] = fmaf(fv, cst[j], iv * gv);
            const float hv = ov * fast_tanh(cst[j]);
            hvv[j] = hv;
            hlast[j] = hv;
        }
        *((v4f*)(out + ((size_t)t * BATCH + eb) * HID + ug)) = hvv;
        *((v4bf*)(hbuf + (size_t)((t + 1) & 1) * HSZ + (size_t)eb * HID + ug)) =
            __builtin_convertvector(hvv, v4bf);

        // ---- publish h for step t (device scope) ----
        __builtin_amdgcn_fence(__ATOMIC_RELEASE, "agent");  // global_wb
        __syncthreads();
        if (tid == 0)
            __hip_atomic_fetch_add(&counters[t], 1, __ATOMIC_RELEASE,
                                   __HIP_MEMORY_SCOPE_AGENT);
    }

    // ---- final hx, cx ----
    float* hx = out + (size_t)SEQ * BATCH * HID;
    float* cx = hx + BATCH * HID;
#pragma unroll
    for (int j = 0; j < 4; ++j) {
        hx[(size_t)eb * HID + ug + j] = hlast[j];
        cx[(size_t)eb * HID + ug + j] = cst[j];
    }
}

// ---------------------------------------------------------------------------
extern "C" void kernel_launch(void* const* d_in, const int* in_sizes, int n_in,
                              void* d_out, int out_size, void* d_ws, size_t ws_size,
                              hipStream_t stream) {
    const float* xin = (const float*)d_in[0];
    const float* Wf  = (const float*)d_in[1];
    const float* bf  = (const float*)d_in[2];
    const float* Wi  = (const float*)d_in[3];
    const float* bi  = (const float*)d_in[4];
    const float* Wg  = (const float*)d_in[5];
    const float* bg  = (const float*)d_in[6];
    const float* Wo  = (const float*)d_in[7];
    const float* bo  = (const float*)d_in[8];

    char*   ws       = (char*)d_ws;
    int*    counters = (int*)(ws + WS_COUNTERS);
    __bf16* hbuf     = (__bf16*)(ws + WS_HBUF);
    __bf16* WT       = (__bf16*)(ws + WS_WT);
    float*  out      = (float*)d_out;

    // zero counters[2048] + hbuf[0] (contiguous 73728 bytes)
    const int n4 = (SEQ * 4 + BATCH * HID * 2) / 16;         // 4608 uint4
    lstm_zero_ws<<<(n4 + 255) / 256, 256, 0, stream>>>((uint4*)ws, n4);

    // transcode weights to bf16 (deterministic, cheap; re-done every call)
    lstm_convert_w<<<(NCOL * KTOT / 4 + 255) / 256, 256, 0, stream>>>(Wf, Wi, Wg, Wo, WT);

    const size_t smem = (size_t)BATCH * ASTR * sizeof(__bf16)   // A stage: 134144
                      + (size_t)BATCH * GSTR * sizeof(float);   // gates:   17408
    lstm_seq_kernel<<<NWG, NTHR, smem, stream>>>(xin, bf, bi, bg, bo,
                                                 WT, hbuf, counters, out);
}
